// GCN_1872605741592
// MI455X (gfx1250) — matrix operand order repaired and checked by reading, not compiled, not run on hardware
//
#include <hip/hip_runtime.h>
#include <hip/hip_bf16.h>

// ---------------------------------------------------------------------------
// 2-layer GCN for MI455X (gfx1250, wave32).
//  h1   = x @ W1                     (WMMA bf16/f32, LDS-staged, async-to-LDS)
//  agg1 = scatter_add(h1[src]*dis[s]*dis[d] -> dst)      (f32 atomics, L2-resident)
//  h1r  = relu(agg1 + h1*dis^2 + b1)
//  h2   = h1r @ W2                   (WMMA bf16, K=16 zero-padded to 32)
//  agg2 = scatter_add(h2[src]*...)
//  out  = log_softmax(agg2 + h2*dis^2 + b2)
// ---------------------------------------------------------------------------

#define F_IN 512
#define HID  16
#define CLS  16

#define A_STRIDE   36   // floats; 144B rows: 16B-aligned b128, conflict-free b64 reads
#define WT_STRIDE 516   // bf16 elems; 1032B rows: 4B-aligned b32, conflict-free reads

typedef __attribute__((ext_vector_type(16))) __bf16 v16bf;
typedef __attribute__((ext_vector_type(2)))  __bf16 bf16x2;
typedef __attribute__((ext_vector_type(8)))  float  v8f;
typedef __attribute__((ext_vector_type(4)))  int    i32x4;

#if __has_builtin(__builtin_amdgcn_global_load_async_to_lds_b128) && \
    __has_builtin(__builtin_amdgcn_s_wait_asynccnt)
#define USE_ASYNC_LDS 1
#else
#define USE_ASYNC_LDS 0
#endif

// builtin prototype (from hipcc diagnostics):
//   void __builtin_amdgcn_global_load_async_to_lds_b128(
//       __attribute__((address_space(1))) i32x4* src,   // global
//       __attribute__((address_space(3))) i32x4* dst,   // LDS
//       int imm_offset, int cpol);
#define TO_GLOBAL_V4(p) ((__attribute__((address_space(1))) i32x4*)(p))
#define TO_LDS_V4(p)    ((__attribute__((address_space(3))) i32x4*)(p))

// ---------------- zero workspace region (vectorized) ----------------
__global__ void zero_kernel(float* p, long long n) {
    long long i = ((long long)blockIdx.x * blockDim.x + threadIdx.x) * 4;
    const long long stride = (long long)gridDim.x * blockDim.x * 4;
    for (; i + 3 < n; i += stride)
        *(float4*)(p + i) = make_float4(0.f, 0.f, 0.f, 0.f);
    if (blockIdx.x == 0 && threadIdx.x == 0)
        for (long long j = n & ~3LL; j < n; ++j) p[j] = 0.f;
}

// ---------------- degree accumulation ----------------
__global__ void degree_kernel(const int* __restrict__ dst, float* __restrict__ deg,
                              long long E) {
    long long e = (long long)blockIdx.x * blockDim.x + threadIdx.x;
    if (e >= E) return;
    atomicAdd(&deg[dst[e]], 1.0f);
}

// dis = rsqrt(deg + 1)  (self loop)
__global__ void finalize_dis_kernel(float* __restrict__ dis, int N) {
    int i = blockIdx.x * blockDim.x + threadIdx.x;
    if (i >= N) return;
    dis[i] = rsqrtf(dis[i] + 1.0f);
}

// ---------------- GEMM1: h1[N,16] = x[N,512] @ W1[512,16] via WMMA bf16 ------
// One wave per 16-row tile; 16 K-steps of K=32. W1 staged transposed (bf16) in
// LDS once per block. A tile staged through double-buffered LDS, fed either by
// async global->LDS loads (ASYNCcnt) or by a register pipeline fallback.
// WMMA register layouts per ISA 7.12.2:
//  A (16-bit 16x32): lane m=l&15, half=l>>4; vgpr j holds K=(j&3)*2+(j>>2)*16+half*8.
//  B (16-bit 32x16): lane n=l&15, half=l>>4; vgpr j holds K=(j&3)*2+(j>>2)*16+half*8 for col n.
//  C: lane col n=l&15; vgpr v is row v+(l>>4)*8.
__global__ __launch_bounds__(256) void gemm1_wmma_kernel(
        const float* __restrict__ x, const float* __restrict__ W1,
        float* __restrict__ h1, int nTiles, int N) {
    __shared__ __bf16 sWt[HID * WT_STRIDE];      // W1 transposed (bf16): [n][k]
    __shared__ float  sA[8][2][16 * A_STRIDE];   // per-wave double-buffered A tile

    const int tid = threadIdx.x;

    // cooperative transpose-load of W1 into LDS as bf16 (one pass per block)
    for (int i4 = tid * 4; i4 < F_IN * HID; i4 += 256 * 4) {
        const float4 w = *(const float4*)(W1 + i4);
        const int k = i4 >> 4;
        const int n = i4 & 15;
        sWt[(n + 0) * WT_STRIDE + k] = (__bf16)w.x;
        sWt[(n + 1) * WT_STRIDE + k] = (__bf16)w.y;
        sWt[(n + 2) * WT_STRIDE + k] = (__bf16)w.z;
        sWt[(n + 3) * WT_STRIDE + k] = (__bf16)w.w;
    }
    __syncthreads();

    const int lane = tid & 31;
    const int wave = tid >> 5;
    const int tile = __builtin_amdgcn_readfirstlane(blockIdx.x * 8 + wave);
    if (tile >= nTiles) return;                  // scalar branch

    const int half = lane >> 4;
    const int mn   = lane & 15;
    const int tileBase = tile * 16;

    // warm the x rows for this tile (emits global_prefetch_b8)
    __builtin_prefetch(x + (long long)(tileBase + mn) * F_IN + half * 256, 0, 3);

    // staging geometry: chunk p covers row p*4 + (lane>>3), cols (lane&7)*4 .. +3
    const int rSt = lane >> 3;          // 0..3
    const int cSt = (lane & 7) * 4;     // 0..28
    int rr0 = tileBase +  0 + rSt; rr0 = rr0 < N ? rr0 : N - 1;
    int rr1 = tileBase +  4 + rSt; rr1 = rr1 < N ? rr1 : N - 1;
    int rr2 = tileBase +  8 + rSt; rr2 = rr2 < N ? rr2 : N - 1;
    int rr3 = tileBase + 12 + rSt; rr3 = rr3 < N ? rr3 : N - 1;
    const float* g0 = x + (long long)rr0 * F_IN + cSt;
    const float* g1 = x + (long long)rr1 * F_IN + cSt;
    const float* g2 = x + (long long)rr2 * F_IN + cSt;
    const float* g3 = x + (long long)rr3 * F_IN + cSt;
    const int l0 = ( 0 + rSt) * A_STRIDE + cSt;
    const int l1 = ( 4 + rSt) * A_STRIDE + cSt;
    const int l2 = ( 8 + rSt) * A_STRIDE + cSt;
    const int l3 = (12 + rSt) * A_STRIDE + cSt;

    v8f c = {};

#if USE_ASYNC_LDS
    // prologue: fill buffer 0 for k0 = 0 (4 async b128 per lane -> ASYNCcnt 4)
    {
        float* b0 = sA[wave][0];
        __builtin_amdgcn_global_load_async_to_lds_b128(TO_GLOBAL_V4(g0), TO_LDS_V4(b0 + l0), 0, 0);
        __builtin_amdgcn_global_load_async_to_lds_b128(TO_GLOBAL_V4(g1), TO_LDS_V4(b0 + l1), 0, 0);
        __builtin_amdgcn_global_load_async_to_lds_b128(TO_GLOBAL_V4(g2), TO_LDS_V4(b0 + l2), 0, 0);
        __builtin_amdgcn_global_load_async_to_lds_b128(TO_GLOBAL_V4(g3), TO_LDS_V4(b0 + l3), 0, 0);
    }
    for (int k0 = 0; k0 < F_IN; k0 += 32) {
        const int cur = (k0 >> 5) & 1;
        const int kN  = (k0 + 32 < F_IN) ? (k0 + 32) : 0;  // harmless tail reload
        float* nb = sA[wave][cur ^ 1];
        __builtin_amdgcn_global_load_async_to_lds_b128(TO_GLOBAL_V4(g0 + kN), TO_LDS_V4(nb + l0), 0, 0);
        __builtin_amdgcn_global_load_async_to_lds_b128(TO_GLOBAL_V4(g1 + kN), TO_LDS_V4(nb + l1), 0, 0);
        __builtin_amdgcn_global_load_async_to_lds_b128(TO_GLOBAL_V4(g2 + kN), TO_LDS_V4(nb + l2), 0, 0);
        __builtin_amdgcn_global_load_async_to_lds_b128(TO_GLOBAL_V4(g3 + kN), TO_LDS_V4(nb + l3), 0, 0);
        // async loads retire in order: <=4 outstanding => current buffer ready
        __builtin_amdgcn_s_wait_asynccnt(4);
        asm volatile("" ::: "memory");
        const float* curA = sA[wave][cur];
        v16bf a, b;
#pragma unroll
        for (int j = 0; j < 8; ++j) {
            const int ka = (j & 3) * 2 + (j >> 2) * 16 + half * 8;
            const float2 av = *(const float2*)(curA + mn * A_STRIDE + ka);
            a[2 * j]     = (__bf16)av.x;
            a[2 * j + 1] = (__bf16)av.y;
            const int kb = k0 + (j & 3) * 2 + (j >> 2) * 16 + half * 8;
            const bf16x2 bv = *(const bf16x2*)(sWt + mn * WT_STRIDE + kb);
            b[2 * j]     = bv.x;
            b[2 * j + 1] = bv.y;
        }
        c = __builtin_amdgcn_wmma_f32_16x16x32_bf16(
                false, a, false, b, (short)0, c, false, false);
    }
#else
    // register-pipelined fallback: named float4s (no spillable array), ds_store
    float* myA = sA[wave][0];
    float4 r0 = *(const float4*)g0;
    float4 r1 = *(const float4*)g1;
    float4 r2 = *(const float4*)g2;
    float4 r3 = *(const float4*)g3;
    for (int k0 = 0; k0 < F_IN; k0 += 32) {
        *(float4*)(myA + l0) = r0;
        *(float4*)(myA + l1) = r1;
        *(float4*)(myA + l2) = r2;
        *(float4*)(myA + l3) = r3;
        const int kN = (k0 + 32 < F_IN) ? (k0 + 32) : 0;   // unconditional reload
        r0 = *(const float4*)(g0 + kN);
        r1 = *(const float4*)(g1 + kN);
        r2 = *(const float4*)(g2 + kN);
        r3 = *(const float4*)(g3 + kN);
        v16bf a, b;
#pragma unroll
        for (int j = 0; j < 8; ++j) {
            const int ka = (j & 3) * 2 + (j >> 2) * 16 + half * 8;
            const float2 av = *(const float2*)(myA + mn * A_STRIDE + ka);
            a[2 * j]     = (__bf16)av.x;
            a[2 * j + 1] = (__bf16)av.y;
            const int kb = k0 + (j & 3) * 2 + (j >> 2) * 16 + half * 8;
            const bf16x2 bv = *(const bf16x2*)(sWt + mn * WT_STRIDE + kb);
            b[2 * j]     = bv.x;
            b[2 * j + 1] = bv.y;
        }
        c = __builtin_amdgcn_wmma_f32_16x16x32_bf16(
                false, a, false, b, (short)0, c, false, false);
    }
#endif

    // store C; scalar fast path (tile is SGPR) avoids exec juggling
    if (tileBase + 16 <= N) {
#pragma unroll
        for (int v = 0; v < 8; ++v)
            h1[(long long)(tileBase + v + half * 8) * HID + mn] = c[v];
    } else {
#pragma unroll
        for (int v = 0; v < 8; ++v) {
            const int r = tileBase + v + half * 8;
            if (r < N) h1[(long long)r * HID + mn] = c[v];
        }
    }
}

// ---------------- GEMM2: h2[N,16] = h1r[N,16] @ W2[16,16], K padded to 32 ----
__global__ void gemm2_wmma_kernel(const float* __restrict__ hin,
                                  const float* __restrict__ W2,
                                  float* __restrict__ hout,
                                  int nTiles, int N) {
    const int lane = threadIdx.x & 31;
    const int wave = threadIdx.x >> 5;
    const int tile = __builtin_amdgcn_readfirstlane(blockIdx.x * (blockDim.x >> 5) + wave);
    if (tile >= nTiles) return;

    const int half = lane >> 4;
    const int mn   = lane & 15;
    const int tileBase = tile * 16;
    int row = tileBase + mn;
    const int arow = row < N ? row : N - 1;

    v16bf a, b;
#pragma unroll
    for (int j = 0; j < 8; ++j) {
        const int ka = (j & 3) * 2 + (j >> 2) * 16 + half * 8;   // 0..31, even
        float a0 = 0.0f, a1 = 0.0f;
        if (ka < HID) {                       // ka<=14 -> ka+1 also valid
            const float2 av = *(const float2*)(hin + (long long)arow * HID + ka);
            a0 = av.x; a1 = av.y;
        }
        a[2 * j]     = (__bf16)a0;
        a[2 * j + 1] = (__bf16)a1;
        const int kb = (j & 3) * 2 + (j >> 2) * 16 + half * 8;     // valid only for j<4
        float b0 = 0.0f, b1 = 0.0f;
        if (kb < HID) {
            b0 = W2[kb * CLS + mn];
            b1 = W2[(kb + 1) * CLS + mn];
        }
        b[2 * j]     = (__bf16)b0;
        b[2 * j + 1] = (__bf16)b1;
    }
    v8f c = {};
    c = __builtin_amdgcn_wmma_f32_16x16x32_bf16(
            false, a, false, b, (short)0, c, false, false);

    if (tileBase + 16 <= N) {
#pragma unroll
        for (int v = 0; v < 8; ++v)
            hout[(long long)(tileBase + v + half * 8) * CLS + mn] = c[v];
    } else {
#pragma unroll
        for (int v = 0; v < 8; ++v) {
            const int r = tileBase + v + half * 8;
            if (r < N) hout[(long long)r * CLS + mn] = c[v];
        }
    }
}

// ---------------- edge aggregation: 4 lanes per edge, float4 gather ----------
// h and agg (6.4 MB each) and dis (400 KB) are L2-resident on MI455X (192 MB L2);
// HBM traffic here is essentially just the streamed edge_index.
__global__ void aggregate_kernel(const int* __restrict__ src,
                                 const int* __restrict__ dst,
                                 const float* __restrict__ dis,
                                 const float* __restrict__ h,
                                 float* __restrict__ agg,
                                 long long E) {
    long long t = (long long)blockIdx.x * blockDim.x + threadIdx.x;
    if (t >= 4 * E) return;
    const long long e = t >> 2;
    const int g = (int)(t & 3);
    const int s = src[e];
    const int d = dst[e];
    const float w = dis[s] * dis[d];
    const float4 hv = *(const float4*)(h + (long long)s * 16 + g * 4);
    float* ap = agg + (long long)d * 16 + g * 4;
    atomicAdd(ap + 0, hv.x * w);
    atomicAdd(ap + 1, hv.y * w);
    atomicAdd(ap + 2, hv.z * w);
    atomicAdd(ap + 3, hv.w * w);
}

// ---------------- finish layer 1: out = relu(agg + h*dis^2 + b) -------------
__global__ void finish1_kernel(const float* __restrict__ agg,
                               const float* __restrict__ h,
                               const float* __restrict__ dis,
                               const float* __restrict__ b,
                               float* __restrict__ out, int N) {
    int i = blockIdx.x * blockDim.x + threadIdx.x;
    if (i >= N) return;
    const float dd = dis[i] * dis[i];
#pragma unroll
    for (int q = 0; q < HID / 4; ++q) {
        const float4 av = *(const float4*)(agg + (long long)i * HID + q * 4);
        const float4 hv = *(const float4*)(h + (long long)i * HID + q * 4);
        const float4 bv = *(const float4*)(b + q * 4);
        float4 o;
        o.x = fmaxf(av.x + hv.x * dd + bv.x, 0.0f);
        o.y = fmaxf(av.y + hv.y * dd + bv.y, 0.0f);
        o.z = fmaxf(av.z + hv.z * dd + bv.z, 0.0f);
        o.w = fmaxf(av.w + hv.w * dd + bv.w, 0.0f);
        *(float4*)(out + (long long)i * HID + q * 4) = o;
    }
}

// ---------------- finish layer 2 + log_softmax ------------------------------
__global__ void finish2_kernel(const float* __restrict__ agg,
                               const float* __restrict__ h,
                               const float* __restrict__ dis,
                               const float* __restrict__ b,
                               float* __restrict__ out, int N) {
    int i = blockIdx.x * blockDim.x + threadIdx.x;
    if (i >= N) return;
    const float dd = dis[i] * dis[i];
    float v[CLS];
#pragma unroll
    for (int q = 0; q < CLS / 4; ++q) {
        const float4 av = *(const float4*)(agg + (long long)i * CLS + q * 4);
        const float4 hv = *(const float4*)(h + (long long)i * CLS + q * 4);
        const float4 bv = *(const float4*)(b + q * 4);
        v[q * 4 + 0] = av.x + hv.x * dd + bv.x;
        v[q * 4 + 1] = av.y + hv.y * dd + bv.y;
        v[q * 4 + 2] = av.z + hv.z * dd + bv.z;
        v[q * 4 + 3] = av.w + hv.w * dd + bv.w;
    }
    float mx = v[0];
#pragma unroll
    for (int f = 1; f < CLS; ++f) mx = fmaxf(mx, v[f]);
    float sum = 0.0f;
#pragma unroll
    for (int f = 0; f < CLS; ++f) sum += __expf(v[f] - mx);
    const float lse = __logf(sum);
#pragma unroll
    for (int f = 0; f < CLS; ++f)
        out[(long long)i * CLS + f] = v[f] - mx - lse;
}

// ---------------------------------------------------------------------------
extern "C" void kernel_launch(void* const* d_in, const int* in_sizes, int n_in,
                              void* d_out, int out_size, void* d_ws, size_t ws_size,
                              hipStream_t stream) {
    const float* x  = (const float*)d_in[0];
    const int*   ei = (const int*)d_in[1];       // [2, E] int32
    const float* W1 = (const float*)d_in[2];
    const float* b1 = (const float*)d_in[3];
    const float* W2 = (const float*)d_in[4];
    const float* b2 = (const float*)d_in[5];
    float* out = (float*)d_out;

    const int       N = in_sizes[0] / F_IN;
    const long long E = (long long)in_sizes[1] / 2;
    const int* src = ei;
    const int* dst = ei + E;

    // workspace layout (floats): dis | agg1 | agg2 | h1 | h1r | h2
    float* ws   = (float*)d_ws;
    float* dis  = ws;
    float* agg1 = dis  + N;
    float* agg2 = agg1 + (size_t)N * HID;
    float* h1   = agg2 + (size_t)N * CLS;
    float* h1r  = h1   + (size_t)N * HID + 32;   // +pad for K-overread safety
    float* h2   = h1r  + (size_t)N * HID + 32;

    const int BS = 256;
    const long long nzero = (long long)N * (1 + HID + CLS);  // dis+agg1+agg2

    zero_kernel<<<2048, BS, 0, stream>>>(ws, nzero);

    degree_kernel<<<(int)((E + BS - 1) / BS), BS, 0, stream>>>(dst, dis, E);
    finalize_dis_kernel<<<(N + BS - 1) / BS, BS, 0, stream>>>(dis, N);

    const int nTiles = (N + 15) / 16;
    const int gemmGrid = (nTiles + 7) / 8;       // 8 waves -> 8 tiles per block

    gemm1_wmma_kernel<<<gemmGrid, BS, 0, stream>>>(x, W1, h1, nTiles, N);

    const long long aggThreads = 4 * E;
    aggregate_kernel<<<(int)((aggThreads + BS - 1) / BS), BS, 0, stream>>>(
        src, dst, dis, h1, agg1, E);

    finish1_kernel<<<(N + BS - 1) / BS, BS, 0, stream>>>(agg1, h1, dis, b1, h1r, N);

    gemm2_wmma_kernel<<<gemmGrid, BS, 0, stream>>>(h1r, W2, h2, nTiles, N);

    aggregate_kernel<<<(int)((aggThreads + BS - 1) / BS), BS, 0, stream>>>(
        src, dst, dis, h2, agg2, E);

    finish2_kernel<<<(N + BS - 1) / BS, BS, 0, stream>>>(agg2, h2, dis, b2, out, N);
}
